// LocationAwareAttention_89154931130797
// MI455X (gfx1250) — matrix operand order repaired and checked by reading, not compiled, not run on hardware
//
#include <hip/hip_runtime.h>
#include <hip/hip_bf16.h>

#define BSZ 2
#define SEQ 2048
#define CH  1024
#define NH  16
#define HDM 64
#define GHD 256
#define BN  (BSZ*SEQ)

typedef _Float16 f16;
typedef __attribute__((ext_vector_type(16))) _Float16 v16h;
typedef __attribute__((ext_vector_type(8)))  _Float16 v8h;
typedef __attribute__((ext_vector_type(8)))  float    v8f;

union AFrag { v16h v; v8h h[2]; };

__device__ __forceinline__ v8f vzero8f() {
    v8f z;
#pragma unroll
    for (int i = 0; i < 8; ++i) z[i] = 0.0f;
    return z;
}

// A-matrix 16x32 f16 fragment from row-major [rows, ld] (ISA 7.12.2 layout):
// lanes 0-15: m=lane, K in {k0+0..7, k0+16..23}; lanes 16-31: m=lane-16, K in {k0+8..15, k0+24..31}
__device__ __forceinline__ v16h load_a_frag(const f16* base, int row0, int k0, int ld, int lane) {
    int hl = lane >> 4, m = lane & 15;
    const f16* p = base + (size_t)(row0 + m) * ld + k0 + 8 * hl;
    AFrag a;
    a.h[0] = *(const v8h*)p;
    a.h[1] = *(const v8h*)(p + 16);
    return a.v;
}

// B-matrix 32x16 f16 fragment where B[k][n] = W[(col0+n)*ld + k0 + k]
// (row-major weight acting as B^T): per-lane two 16B loads.
// half e <-> K = 8*hl + e (e<8), 16 + 8*hl + (e-8) (e>=8), col n = lane&15.
__device__ __forceinline__ v16h load_bT_frag(const f16* W, int col0, int k0, int ld, int lane) {
    int hl = lane >> 4, n = lane & 15;
    const f16* p = W + (size_t)(col0 + n) * ld + k0 + 8 * hl;
    AFrag b;
    b.h[0] = *(const v8h*)p;
    b.h[1] = *(const v8h*)(p + 16);
    return b.v;
}

__global__ void f32_to_f16_kernel(const float* __restrict__ src, f16* __restrict__ dst, int n) {
    int i = blockIdx.x * blockDim.x + threadIdx.x;
    int stride = gridDim.x * blockDim.x;
    for (; i < n; i += stride) dst[i] = (f16)src[i];
}

__global__ void gate_kernel(const float* __restrict__ mask,
                            const float* __restrict__ g1w, const float* __restrict__ g1b,
                            const float* __restrict__ g2w, const float* __restrict__ g2b,
                            float* __restrict__ gate) {
    int i = blockIdx.x * blockDim.x + threadIdx.x;
    if (i >= BN) return;
    float m = mask[i];
    float acc = 0.f;
    for (int j = 0; j < GHD; ++j) {
        float hv = fmaxf(m * g1w[j] + g1b[j], 0.f);
        acc += hv * g2w[j];
    }
    float x = acc + g2b[0];
    gate[i] = 1.f / (1.f + __expf(-x));
}

// qkv = xh[BN,CH] @ qkv_w^T[CH,3CH] + bias.
// One wave computes a 32x64 strip, software-pipelined over K.
// Q pre-scaled by HD^-0.5; Q,K stored [B,NH,SEQ,HDM]; V stored transposed [B,NH,HDM,SEQ].
__global__ void __launch_bounds__(256) qkv_gemm_kernel(const f16* __restrict__ xh,
                                                       const f16* __restrict__ wh,
                                                       const float* __restrict__ bias,
                                                       f16* __restrict__ qb,
                                                       f16* __restrict__ kb,
                                                       f16* __restrict__ vt) {
    int lane = threadIdx.x & 31;
    int wid  = (blockIdx.x * blockDim.x + threadIdx.x) >> 5;
    const int MT = BN / 32;          // 128 row tiles
    int mtile  = wid % MT;
    int nstrip = wid / MT;           // 0..47
    int m0 = mtile * 32;
    int n0 = nstrip * 64;

    v8f acc[2][4];
#pragma unroll
    for (int s = 0; s < 2; ++s)
#pragma unroll
        for (int i = 0; i < 4; ++i) acc[s][i] = vzero8f();

    v16h a0 = load_a_frag(xh, m0,      0, CH, lane);
    v16h a1 = load_a_frag(xh, m0 + 16, 0, CH, lane);
    v16h b0 = load_bT_frag(wh, n0,      0, CH, lane);
    v16h b1 = load_bT_frag(wh, n0 + 16, 0, CH, lane);
    v16h b2 = load_bT_frag(wh, n0 + 32, 0, CH, lane);
    v16h b3 = load_bT_frag(wh, n0 + 48, 0, CH, lane);

    for (int k0 = 0; k0 < CH; k0 += 32) {
        int kn = (k0 + 32 < CH) ? (k0 + 32) : k0;   // clamp: last iter reloads (unused)
        // prefetch A stream two chunks ahead
        {
            int kp = (k0 + 64 < CH) ? (k0 + 64) : (CH - 32);
            const f16* pp = xh + (size_t)(m0 + (lane & 15)) * CH + kp;
            __builtin_prefetch(pp, 0, 3);
        }
        // issue next chunk's loads before consuming current fragments
        v16h na0 = load_a_frag(xh, m0,      kn, CH, lane);
        v16h na1 = load_a_frag(xh, m0 + 16, kn, CH, lane);
        v16h nb0 = load_bT_frag(wh, n0,      kn, CH, lane);
        v16h nb1 = load_bT_frag(wh, n0 + 16, kn, CH, lane);
        v16h nb2 = load_bT_frag(wh, n0 + 32, kn, CH, lane);
        v16h nb3 = load_bT_frag(wh, n0 + 48, kn, CH, lane);

        acc[0][0] = __builtin_amdgcn_wmma_f32_16x16x32_f16(false, a0, false, b0, (short)0, acc[0][0], false, false);
        acc[0][1] = __builtin_amdgcn_wmma_f32_16x16x32_f16(false, a0, false, b1, (short)0, acc[0][1], false, false);
        acc[0][2] = __builtin_amdgcn_wmma_f32_16x16x32_f16(false, a0, false, b2, (short)0, acc[0][2], false, false);
        acc[0][3] = __builtin_amdgcn_wmma_f32_16x16x32_f16(false, a0, false, b3, (short)0, acc[0][3], false, false);
        acc[1][0] = __builtin_amdgcn_wmma_f32_16x16x32_f16(false, a1, false, b0, (short)0, acc[1][0], false, false);
        acc[1][1] = __builtin_amdgcn_wmma_f32_16x16x32_f16(false, a1, false, b1, (short)0, acc[1][1], false, false);
        acc[1][2] = __builtin_amdgcn_wmma_f32_16x16x32_f16(false, a1, false, b2, (short)0, acc[1][2], false, false);
        acc[1][3] = __builtin_amdgcn_wmma_f32_16x16x32_f16(false, a1, false, b3, (short)0, acc[1][3], false, false);

        a0 = na0; a1 = na1; b0 = nb0; b1 = nb1; b2 = nb2; b3 = nb3;
    }

    int hl = lane >> 4, nl = lane & 15;
#pragma unroll
    for (int nt = 0; nt < 4; ++nt) {
        int c = n0 + nt * 16 + nl;
        float bv = bias[c];
        int which = c >> 10;               // 0=q 1=k 2=v
        int cc = c & (CH - 1);
        int h = cc >> 6, d = cc & (HDM - 1);
        float scale = (which == 0) ? 0.125f : 1.0f;
#pragma unroll
        for (int sub = 0; sub < 2; ++sub) {
#pragma unroll
            for (int r = 0; r < 8; ++r) {
                int mg = m0 + sub * 16 + r + 8 * hl;
                int b_ = mg >> 11;             // / SEQ
                int n_ = mg & (SEQ - 1);
                int bh = b_ * NH + h;
                float val = (acc[sub][nt][r] + bv) * scale;
                if (which == 2) {
                    vt[((size_t)bh * HDM + d) * SEQ + n_] = (f16)val;     // transposed V
                } else {
                    f16* dst = (which == 0) ? qb : kb;
                    dst[((size_t)bh * SEQ + n_) * HDM + d] = (f16)val;
                }
            }
        }
    }
}

// Flash attention: one wave per (b,h,16-query tile); online softmax over key chunks of 32.
// K frags pipelined one chunk ahead; V (transposed layout) frags issued before softmax.
__global__ void __launch_bounds__(256) attn_kernel(const f16* __restrict__ qb,
                                                   const f16* __restrict__ kb,
                                                   const f16* __restrict__ vt,
                                                   const float* __restrict__ gate,
                                                   f16* __restrict__ attn_out) {
    __shared__ __align__(16) f16 lds[8][16 * 32];   // per-wave 16x32 P transpose tile

    int lane = threadIdx.x & 31;
    int wblk = threadIdx.x >> 5;
    int wid  = (blockIdx.x * blockDim.x + threadIdx.x) >> 5;
    int qtile = wid & (SEQ / 16 - 1);   // 0..127
    int bh    = wid >> 7;               // 0..31
    int b = bh >> 4, h = bh & 15;

    const f16* Q  = qb + (size_t)bh * SEQ * HDM;
    const f16* K  = kb + (size_t)bh * SEQ * HDM;
    const f16* Vt = vt + (size_t)bh * HDM * SEQ;    // [HDM, SEQ]
    const float* g = gate + b * SEQ;

    int q0 = qtile * 16;
    int hl = lane >> 4, nl = lane & 15;

    v16h aq0 = load_a_frag(Q, q0, 0,  HDM, lane);
    v16h aq1 = load_a_frag(Q, q0, 32, HDM, lane);

    v8f o[4];
#pragma unroll
    for (int i = 0; i < 4; ++i) o[i] = vzero8f();
    float mrow[8], lrow[8];
#pragma unroll
    for (int r = 0; r < 8; ++r) { mrow[r] = -1e30f; lrow[r] = 0.f; }

    // preload K fragments for chunk 0: [tile(2)][hd-chunk(2)]
    v16h kc0 = load_bT_frag(K, 0,  0,  HDM, lane);
    v16h kc1 = load_bT_frag(K, 0,  32, HDM, lane);
    v16h kc2 = load_bT_frag(K, 16, 0,  HDM, lane);
    v16h kc3 = load_bT_frag(K, 16, 32, HDM, lane);

    for (int c0 = 0; c0 < SEQ; c0 += 32) {
        // V fragments for the current chunk (two 16B loads per lane each)
        v16h vf[4];
#pragma unroll
        for (int dt = 0; dt < 4; ++dt)
            vf[dt] = load_bT_frag(Vt, dt * 16, c0, SEQ, lane);
        // next chunk's K fragments (clamped; last iter reloads, unused)
        int cn = (c0 + 32 < SEQ) ? (c0 + 32) : c0;
        v16h nk0 = load_bT_frag(K, cn,      0,  HDM, lane);
        v16h nk1 = load_bT_frag(K, cn,      32, HDM, lane);
        v16h nk2 = load_bT_frag(K, cn + 16, 0,  HDM, lane);
        v16h nk3 = load_bT_frag(K, cn + 16, 32, HDM, lane);
        float g0 = g[c0 + nl];
        float g1 = g[c0 + 16 + nl];

        // S = Q * K^T for two 16-key column tiles (contract over HD=64 in 2 chunks)
        v8f s[2];
        {
            v8f z = vzero8f();
            z = __builtin_amdgcn_wmma_f32_16x16x32_f16(false, aq0, false, kc0, (short)0, z, false, false);
            z = __builtin_amdgcn_wmma_f32_16x16x32_f16(false, aq1, false, kc1, (short)0, z, false, false);
            s[0] = z;
            z = vzero8f();
            z = __builtin_amdgcn_wmma_f32_16x16x32_f16(false, aq0, false, kc2, (short)0, z, false, false);
            z = __builtin_amdgcn_wmma_f32_16x16x32_f16(false, aq1, false, kc3, (short)0, z, false, false);
            s[1] = z;
        }
        // per-key gate (Q already carries the 1/sqrt(HD) scale)
#pragma unroll
        for (int r = 0; r < 8; ++r) { s[0][r] *= g0; s[1][r] *= g1; }

        // online softmax: each row's 16 values live across 16 lanes of one acc element
#pragma unroll
        for (int r = 0; r < 8; ++r) {
            float vmax = fmaxf(s[0][r], s[1][r]);
            vmax = fmaxf(vmax, __shfl_xor(vmax, 1));
            vmax = fmaxf(vmax, __shfl_xor(vmax, 2));
            vmax = fmaxf(vmax, __shfl_xor(vmax, 4));
            vmax = fmaxf(vmax, __shfl_xor(vmax, 8));
            float mnew = fmaxf(mrow[r], vmax);
            float corr = __expf(mrow[r] - mnew);
            mrow[r] = mnew;
            float p0 = __expf(s[0][r] - mnew);
            float p1 = __expf(s[1][r] - mnew);
            s[0][r] = p0; s[1][r] = p1;
            float ps = p0 + p1;
            ps += __shfl_xor(ps, 1);
            ps += __shfl_xor(ps, 2);
            ps += __shfl_xor(ps, 4);
            ps += __shfl_xor(ps, 8);
            lrow[r] = lrow[r] * corr + ps;
#pragma unroll
            for (int dt = 0; dt < 4; ++dt) o[dt][r] *= corr;
        }

        // transpose P (C-layout -> A-layout) through LDS
#pragma unroll
        for (int r = 0; r < 8; ++r) {
            int m = r + 8 * hl;
            lds[wblk][m * 32 + nl]      = (f16)s[0][r];
            lds[wblk][m * 32 + 16 + nl] = (f16)s[1][r];
        }
        asm volatile("s_wait_dscnt 0x0" ::: "memory");

        AFrag ap;
        {
            const f16* p = &lds[wblk][nl * 32 + 8 * hl];
            ap.h[0] = *(const v8h*)p;
            ap.h[1] = *(const v8h*)(p + 16);
        }

        // O += P * V  (4 dim-tiles of 16)
#pragma unroll
        for (int dt = 0; dt < 4; ++dt) {
            o[dt] = __builtin_amdgcn_wmma_f32_16x16x32_f16(false, ap.v, false, vf[dt],
                                                           (short)0, o[dt], false, false);
        }

        kc0 = nk0; kc1 = nk1; kc2 = nk2; kc3 = nk3;
    }

    // epilogue: divide by row sums, store [B, SEQ, CH] f16
#pragma unroll
    for (int r = 0; r < 8; ++r) {
        float inv = 1.0f / lrow[r];
        int qg = q0 + r + 8 * hl;
        size_t rowoff = ((size_t)(b * SEQ + qg)) * CH + h * HDM;
#pragma unroll
        for (int dt = 0; dt < 4; ++dt) {
            attn_out[rowoff + dt * 16 + nl] = (f16)(o[dt][r] * inv);
        }
    }
}

// out[BN,CH] f32 = attn_out[BN,CH] f16 @ proj_w^T + proj_b ; 32x64 per-wave, pipelined.
__global__ void __launch_bounds__(256) proj_gemm_kernel(const f16* __restrict__ ah,
                                                        const f16* __restrict__ wh,
                                                        const float* __restrict__ bias,
                                                        float* __restrict__ out) {
    int lane = threadIdx.x & 31;
    int wid  = (blockIdx.x * blockDim.x + threadIdx.x) >> 5;
    const int MT = BN / 32;
    int mtile  = wid % MT;
    int nstrip = wid / MT;          // 0..15
    int m0 = mtile * 32;
    int n0 = nstrip * 64;

    v8f acc[2][4];
#pragma unroll
    for (int s = 0; s < 2; ++s)
#pragma unroll
        for (int i = 0; i < 4; ++i) acc[s][i] = vzero8f();

    v16h a0 = load_a_frag(ah, m0,      0, CH, lane);
    v16h a1 = load_a_frag(ah, m0 + 16, 0, CH, lane);
    v16h b0 = load_bT_frag(wh, n0,      0, CH, lane);
    v16h b1 = load_bT_frag(wh, n0 + 16, 0, CH, lane);
    v16h b2 = load_bT_frag(wh, n0 + 32, 0, CH, lane);
    v16h b3 = load_bT_frag(wh, n0 + 48, 0, CH, lane);

    for (int k0 = 0; k0 < CH; k0 += 32) {
        int kn = (k0 + 32 < CH) ? (k0 + 32) : k0;
        {
            int kp = (k0 + 64 < CH) ? (k0 + 64) : (CH - 32);
            const f16* pp = ah + (size_t)(m0 + (lane & 15)) * CH + kp;
            __builtin_prefetch(pp, 0, 3);
        }
        v16h na0 = load_a_frag(ah, m0,      kn, CH, lane);
        v16h na1 = load_a_frag(ah, m0 + 16, kn, CH, lane);
        v16h nb0 = load_bT_frag(wh, n0,      kn, CH, lane);
        v16h nb1 = load_bT_frag(wh, n0 + 16, kn, CH, lane);
        v16h nb2 = load_bT_frag(wh, n0 + 32, kn, CH, lane);
        v16h nb3 = load_bT_frag(wh, n0 + 48, kn, CH, lane);

        acc[0][0] = __builtin_amdgcn_wmma_f32_16x16x32_f16(false, a0, false, b0, (short)0, acc[0][0], false, false);
        acc[0][1] = __builtin_amdgcn_wmma_f32_16x16x32_f16(false, a0, false, b1, (short)0, acc[0][1], false, false);
        acc[0][2] = __builtin_amdgcn_wmma_f32_16x16x32_f16(false, a0, false, b2, (short)0, acc[0][2], false, false);
        acc[0][3] = __builtin_amdgcn_wmma_f32_16x16x32_f16(false, a0, false, b3, (short)0, acc[0][3], false, false);
        acc[1][0] = __builtin_amdgcn_wmma_f32_16x16x32_f16(false, a1, false, b0, (short)0, acc[1][0], false, false);
        acc[1][1] = __builtin_amdgcn_wmma_f32_16x16x32_f16(false, a1, false, b1, (short)0, acc[1][1], false, false);
        acc[1][2] = __builtin_amdgcn_wmma_f32_16x16x32_f16(false, a1, false, b2, (short)0, acc[1][2], false, false);
        acc[1][3] = __builtin_amdgcn_wmma_f32_16x16x32_f16(false, a1, false, b3, (short)0, acc[1][3], false, false);

        a0 = na0; a1 = na1; b0 = nb0; b1 = nb1; b2 = nb2; b3 = nb3;
    }

    int hl = lane >> 4, nl = lane & 15;
#pragma unroll
    for (int nt = 0; nt < 4; ++nt) {
        int c = n0 + nt * 16 + nl;
        float bv = bias[c];
#pragma unroll
        for (int sub = 0; sub < 2; ++sub) {
#pragma unroll
            for (int r = 0; r < 8; ++r) {
                int mg = m0 + sub * 16 + r + 8 * hl;
                out[(size_t)mg * CH + c] = acc[sub][nt][r] + bv;
            }
        }
    }
}

extern "C" void kernel_launch(void* const* d_in, const int* in_sizes, int n_in,
                              void* d_out, int out_size, void* d_ws, size_t ws_size,
                              hipStream_t stream) {
    (void)in_sizes; (void)n_in; (void)out_size; (void)ws_size;
    const float* x      = (const float*)d_in[0];
    const float* smask  = (const float*)d_in[1];
    const float* qkv_w  = (const float*)d_in[2];
    const float* qkv_b  = (const float*)d_in[3];
    const float* proj_w = (const float*)d_in[4];
    const float* proj_b = (const float*)d_in[5];
    const float* g1_w   = (const float*)d_in[6];
    const float* g1_b   = (const float*)d_in[7];
    const float* g2_w   = (const float*)d_in[8];
    const float* g2_b   = (const float*)d_in[9];
    float* out = (float*)d_out;

    // workspace layout (all slices 256B-aligned)
    char* ws = (char*)d_ws;
    size_t off = 0;
    f16* xh      = (f16*)(ws + off); off += (size_t)BN * CH * 2;          // 8 MB
    f16* qkvwh   = (f16*)(ws + off); off += (size_t)3 * CH * CH * 2;      // 6 MB
    f16* projwh  = (f16*)(ws + off); off += (size_t)CH * CH * 2;          // 2 MB
    f16* qb      = (f16*)(ws + off); off += (size_t)BN * CH * 2;          // 8 MB
    f16* kb      = (f16*)(ws + off); off += (size_t)BN * CH * 2;          // 8 MB
    f16* vt      = (f16*)(ws + off); off += (size_t)BN * CH * 2;          // 8 MB (transposed V)
    float* gateb = (float*)(ws + off); off += (size_t)BN * 4;             // 16 KB
    f16* atto    = (f16*)(ws + off); off += (size_t)BN * CH * 2;          // 8 MB

    // 1) convert activations + weights to f16
    {
        int n = BN * CH;
        f32_to_f16_kernel<<<(n + 2047) / 2048, 256, 0, stream>>>(x, xh, n);
        n = 3 * CH * CH;
        f32_to_f16_kernel<<<(n + 2047) / 2048, 256, 0, stream>>>(qkv_w, qkvwh, n);
        n = CH * CH;
        f32_to_f16_kernel<<<(n + 2047) / 2048, 256, 0, stream>>>(proj_w, projwh, n);
    }

    // 2) per-key gates
    gate_kernel<<<(BN + 255) / 256, 256, 0, stream>>>(smask, g1_w, g1_b, g2_w, g2_b, gateb);

    // 3) QKV projection: (BN/32) * (3CH/64) = 6144 waves -> 768 blocks
    {
        int waves = (BN / 32) * (3 * CH / 64);
        qkv_gemm_kernel<<<waves / 8, 256, 0, stream>>>(xh, qkvwh, qkv_b, qb, kb, vt);
    }

    // 4) flash attention: B*H*(SEQ/16) = 4096 waves -> 512 blocks
    {
        int waves = BSZ * NH * (SEQ / 16);
        attn_kernel<<<waves / 8, 256, 0, stream>>>(qb, kb, vt, gateb, atto);
    }

    // 5) output projection: (BN/32)*(CH/64) = 2048 waves -> 256 blocks
    {
        int waves = (BN / 32) * (CH / 64);
        proj_gemm_kernel<<<waves / 8, 256, 0, stream>>>(atto, projwh, proj_b, out);
    }
}
